// LinkPredictor_90658169684724
// MI455X (gfx1250) — matrix operand order repaired and checked by reading, not compiled, not run on hardware
//
#include <hip/hip_runtime.h>
#include <hip/hip_bf16.h>
#include <math.h>

#define DEVINL __device__ __forceinline__

typedef __attribute__((ext_vector_type(16))) __bf16 bf16x16;
typedef __attribute__((ext_vector_type(8)))  float  f32x8;

static constexpr int NN   = 20000;   // nodes
static constexpr int EE   = 320000;  // edges
static constexpr int ELN  = 100000;  // label edges
static constexpr int INF_ = 256;     // input feat
static constexpr int HIDC = 128;     // hidden per head
static constexpr int OUTC = 128;     // layer2 out
static constexpr int HDS  = 4;       // heads layer1
static constexpr int E2   = EE + NN; // edges + self loops

DEVINL __bf16 f2bf(float f) { return (__bf16)f; }   // native v_cvt (RTNE)

// gelu(tanh approx) = x * sigmoid(1.59577*x + 0.0713548*x^3)
DEVINL float gelu_tanh(float x) {
  float t = 1.5957691216057308f * x + 0.07135481627375633f * x * x * x;
  return x / (1.0f + __expf(-t));
}

// monotone float <-> uint mapping so atomicMax(uint) == float max
DEVINL unsigned fmap_ord(float f) {
  unsigned b = __builtin_bit_cast(unsigned, f);
  return (b & 0x80000000u) ? ~b : (b | 0x80000000u);
}
DEVINL float funmap_ord(unsigned u) {
  unsigned b = (u & 0x80000000u) ? (u & 0x7FFFFFFFu) : ~u;
  return __builtin_bit_cast(float, b);
}

// ---------------------------------------------------------------------------
// Pack fp32 weight matrix B[K,Nn] into WMMA-B-fragment order, bf16:
// P[((kt*ntiles + nt)*32 + lane)*16 + e] = bf16(B[(kt*32 + (lane>>4)*8 + (e&7)
//                                  + (e>>3)*16)*Nn + nt*16 + (lane&15)])
// ---------------------------------------------------------------------------
__global__ void pack_b_kernel(const float* __restrict__ B,
                              __bf16* __restrict__ P, int K, int Nn) {
  int i = blockIdx.x * blockDim.x + threadIdx.x;
  if (i >= K * Nn) return;
  int e    = i & 15;
  int lane = (i >> 4) & 31;
  int tile = i >> 9;
  int ntiles = Nn >> 4;
  int kt = tile / ntiles, nt = tile - kt * ntiles;
  int k = kt * 32 + (lane >> 4) * 8 + (e & 7) + (e >> 3) * 16;
  int n = nt * 16 + (lane & 15);
  P[i] = f2bf(B[(size_t)k * Nn + n]);
}

// ---------------------------------------------------------------------------
// GEMM: C[M,Nn] = A[M,K] @ B + bias.  One wave per 16x64 strip of C: the A
// fragment is loaded once per k-step and reused across 4 WMMAs on packed-bf16
// B fragments (single contiguous 32B load per lane each).
// M%16==0, Nn%64==0, K%32==0.
// ---------------------------------------------------------------------------
__global__ void gemm_bf16wmma_kernel(const float* __restrict__ A,
                                     const __bf16* __restrict__ BP,
                                     const float* __restrict__ bias,
                                     float* __restrict__ C,
                                     int M, int Nn, int K) {
  int wid  = (blockIdx.x * blockDim.x + threadIdx.x) >> 5;
  int lane = threadIdx.x & 31;
  int nstrips = Nn >> 6;                   // strips of 64 columns
  int strips  = (M >> 4) * nstrips;
  if (wid >= strips) return;
  int mt = wid / nstrips, ns = wid - mt * nstrips;
  int m0 = mt << 4, n0 = ns << 6;
  int mlo = lane & 15;
  int khalf = lane >> 4;                   // 0 or 1
  int ntiles = Nn >> 4;

  f32x8 acc0 = {}, acc1 = {}, acc2 = {}, acc3 = {};
  const float* parow = A + (size_t)(m0 + mlo) * K + khalf * 8;
  for (int k0 = 0; k0 < K; k0 += 32) {
    // A fragment: lane group 0: K = k0+[0..7],[16..23]; group 1: +8
    const float* pa = parow + k0;
    bf16x16 af;
#pragma unroll
    for (int e = 0; e < 8; ++e) af[e]     = f2bf(pa[e]);
#pragma unroll
    for (int e = 0; e < 8; ++e) af[8 + e] = f2bf(pa[16 + e]);
    int kt = k0 >> 5;
    const __bf16* pb =
        BP + ((size_t)(kt * ntiles + (ns << 2)) * 32 + lane) * 16;
    bf16x16 b0 = *(const bf16x16*)(pb);
    bf16x16 b1 = *(const bf16x16*)(pb + 512);
    bf16x16 b2 = *(const bf16x16*)(pb + 1024);
    bf16x16 b3 = *(const bf16x16*)(pb + 1536);
    acc0 = __builtin_amdgcn_wmma_f32_16x16x32_bf16(false, af, false, b0, (short)0, acc0, false, false);
    acc1 = __builtin_amdgcn_wmma_f32_16x16x32_bf16(false, af, false, b1, (short)0, acc1, false, false);
    acc2 = __builtin_amdgcn_wmma_f32_16x16x32_bf16(false, af, false, b2, (short)0, acc2, false, false);
    acc3 = __builtin_amdgcn_wmma_f32_16x16x32_bf16(false, af, false, b3, (short)0, acc3, false, false);
  }
  int mb = m0 + (khalf << 3);
  f32x8 accs[4] = {acc0, acc1, acc2, acc3};
#pragma unroll
  for (int j = 0; j < 4; ++j) {
    int nc = n0 + j * 16 + mlo;
    float bv = bias[nc];
#pragma unroll
    for (int r = 0; r < 8; ++r)
      C[(size_t)(mb + r) * Nn + nc] = accs[j][r] + bv;
  }
}

// ---------------------------------------------------------------------------
// Elementwise helpers
// ---------------------------------------------------------------------------
__global__ void fill_f32_kernel(float* p, float v, int n) {
  int i = blockIdx.x * blockDim.x + threadIdx.x;
  if (i < n) p[i] = v;
}
__global__ void fill_u32_kernel(unsigned* p, unsigned v, int n) {
  int i = blockIdx.x * blockDim.x + threadIdx.x;
  if (i < n) p[i] = v;
}
__global__ void init_bias_kernel(float* __restrict__ out,
                                 const float* __restrict__ bias,
                                 int rows, int cols) {
  int i = blockIdx.x * blockDim.x + threadIdx.x;
  if (i < rows * cols) out[i] = bias[i % cols];
}
__global__ void gelu_kernel(float* p, int n) {
  int i = blockIdx.x * blockDim.x + threadIdx.x;
  if (i < n) p[i] = gelu_tanh(p[i]);
}

// ---------------------------------------------------------------------------
// Attention kernels (128 channels per head)
// ---------------------------------------------------------------------------
// wave per (edge, head): logit = att . leaky_relu(xr[dst]+xl[src]); seg max
__global__ void edge_logits_kernel(const float* __restrict__ xl,
                                   const float* __restrict__ xr,
                                   const int* __restrict__ src,
                                   const int* __restrict__ dst,
                                   const float* __restrict__ att,
                                   float* __restrict__ logits,
                                   unsigned* __restrict__ segmax,
                                   int heads) {
  int wid  = (blockIdx.x * blockDim.x + threadIdx.x) >> 5;
  int lane = threadIdx.x & 31;
  if (wid >= E2 * heads) return;
  int e = wid / heads, h = wid - e * heads;
  int s, d;
  if (e < EE) { s = src[e]; d = dst[e]; } else { s = d = e - EE; }
  const float* pl = xl + ((size_t)s * heads + h) * 128;
  const float* pr = xr + ((size_t)d * heads + h) * 128;
  const float* pa = att + (size_t)h * 128;
  float sum = 0.f;
#pragma unroll
  for (int c = 0; c < 4; ++c) {
    int cc = lane + 32 * c;
    float v = pl[cc] + pr[cc];
    v = (v > 0.f) ? v : 0.2f * v;         // leaky_relu, slope 0.2
    sum += v * pa[cc];
  }
#pragma unroll
  for (int m = 16; m; m >>= 1) sum += __shfl_xor(sum, m, 32);
  if (lane == 0) {
    logits[wid] = sum;
    atomicMax(&segmax[(size_t)d * heads + h], fmap_ord(sum));
  }
}

// thread per (edge, head): a = exp(logit - max[dst]); seg sum; a in-place
__global__ void edge_exp_kernel(float* __restrict__ logits,
                                const unsigned* __restrict__ segmax,
                                float* __restrict__ segsum,
                                const int* __restrict__ dst,
                                int heads) {
  int wid = blockIdx.x * blockDim.x + threadIdx.x;
  if (wid >= E2 * heads) return;
  int e = wid / heads, h = wid - e * heads;
  int d = (e < EE) ? dst[e] : (e - EE);
  float a = __expf(logits[wid] - funmap_ord(segmax[(size_t)d * heads + h]));
  logits[wid] = a;
  atomicAdd(&segsum[(size_t)d * heads + h], a);
}

// wave per (edge, head): out[dst] += (a / s[dst]) * xl[src]
__global__ void edge_aggregate_kernel(const float* __restrict__ xl,
                                      const float* __restrict__ alpha,
                                      const float* __restrict__ segsum,
                                      const int* __restrict__ src,
                                      const int* __restrict__ dst,
                                      float* __restrict__ out,
                                      int heads) {
  int wid  = (blockIdx.x * blockDim.x + threadIdx.x) >> 5;
  int lane = threadIdx.x & 31;
  if (wid >= E2 * heads) return;
  int e = wid / heads, h = wid - e * heads;
  int s, d;
  if (e < EE) { s = src[e]; d = dst[e]; } else { s = d = e - EE; }
  float an = alpha[wid] / segsum[(size_t)d * heads + h];
  const float* pl = xl + ((size_t)s * heads + h) * 128;
  float* po = out + ((size_t)d * heads + h) * 128;
#pragma unroll
  for (int c = 0; c < 4; ++c) {
    int cc = lane + 32 * c;
    atomicAdd(&po[cc], an * pl[cc]);
  }
}

// ---------------------------------------------------------------------------
// Fused decoder: out[e] = gelu(concat(z[i],z[j]) @ Wd1 + bd1) @ Wd2 + bd2
// Block = 16 rows; wave w handles hidden-column tile n0 = 16*w (packed Wd1);
// after GELU each wave dots its tile with Wd2 and atomically adds per-row
// partials.  out must be pre-filled with bd2.
// ---------------------------------------------------------------------------
__global__ void decoder_kernel(const float* __restrict__ z2,
                               const int* __restrict__ idx0,
                               const int* __restrict__ idx1,
                               const __bf16* __restrict__ Wd1P,
                               const float* __restrict__ bd1,
                               const float* __restrict__ Wd2,
                               float* __restrict__ out) {
  int lane  = threadIdx.x & 31;
  int w     = threadIdx.x >> 5;   // 0..7 -> column tile
  int m0    = blockIdx.x * 16;
  int mlo   = lane & 15;
  int khalf = lane >> 4;
  int m = m0 + mlo;
  int r0 = idx0[m], r1 = idx1[m];

  f32x8 acc = {};
  for (int k0 = 0; k0 < 256; k0 += 32) {
    // concat(z[i], z[j]) row: fragments never straddle k=128
    const float* arow = (k0 < 128) ? (z2 + (size_t)r0 * 128 + k0)
                                   : (z2 + (size_t)r1 * 128 + (k0 - 128));
    const float* pa = arow + khalf * 8;
    bf16x16 af;
#pragma unroll
    for (int e = 0; e < 8; ++e) af[e]     = f2bf(pa[e]);
#pragma unroll
    for (int e = 0; e < 8; ++e) af[8 + e] = f2bf(pa[16 + e]);
    int kt = k0 >> 5;
    bf16x16 bfv = *(const bf16x16*)(Wd1P + ((size_t)(kt * 8 + w) * 32 + lane) * 16);
    acc = __builtin_amdgcn_wmma_f32_16x16x32_bf16(false, af, false, bfv,
                                                  (short)0, acc, false, false);
  }
  int nc = w * 16 + mlo;
  float wn = Wd2[nc];
  float bn = bd1[nc];
  float ps[8];
#pragma unroll
  for (int r = 0; r < 8; ++r) ps[r] = gelu_tanh(acc[r] + bn) * wn;
  // reduce over the 16 columns within each 16-lane half
#pragma unroll
  for (int msk = 1; msk < 16; msk <<= 1)
#pragma unroll
    for (int r = 0; r < 8; ++r) ps[r] += __shfl_xor(ps[r], msk, 32);
  if (mlo == 0) {
    int mb = m0 + khalf * 8;
#pragma unroll
    for (int r = 0; r < 8; ++r) atomicAdd(&out[mb + r], ps[r]);
  }
}

// ---------------------------------------------------------------------------
extern "C" void kernel_launch(void* const* d_in, const int* in_sizes, int n_in,
                              void* d_out, int out_size, void* d_ws, size_t ws_size,
                              hipStream_t stream) {
  const float* x    = (const float*)d_in[0];
  const int*   ei   = (const int*)d_in[1];     // [2, E]
  const int*   eli  = (const int*)d_in[2];     // [2, EL]
  const float* Wl1  = (const float*)d_in[3];
  const float* bl1  = (const float*)d_in[4];
  const float* Wr1  = (const float*)d_in[5];
  const float* br1  = (const float*)d_in[6];
  const float* att1 = (const float*)d_in[7];
  const float* bias1= (const float*)d_in[8];
  const float* Wl2  = (const float*)d_in[9];
  const float* bl2  = (const float*)d_in[10];
  const float* Wr2  = (const float*)d_in[11];
  const float* br2  = (const float*)d_in[12];
  const float* att2 = (const float*)d_in[13];
  const float* bias2= (const float*)d_in[14];
  const float* Wd1  = (const float*)d_in[15];
  const float* bd1  = (const float*)d_in[16];
  const float* Wd2  = (const float*)d_in[17];
  const float* bd2  = (const float*)d_in[18];
  float* out = (float*)d_out;

  const int* src = ei;
  const int* dst = ei + EE;
  const int* li0 = eli;
  const int* li1 = eli + ELN;

  // workspace layout (floats); layer-2 buffers alias the layer-1 xl region
  float* ws = (float*)d_ws;
  constexpr size_t SZ1 = (size_t)NN * (HDS * HIDC);          // 10.24M floats
  float*    xl1  = ws;
  float*    xr1  = ws + SZ1;
  float*    z1   = ws + 2 * SZ1;
  float*    alph = ws + 3 * SZ1;                             // E2*HDS
  unsigned* smax = (unsigned*)(ws + 3 * SZ1 + (size_t)E2 * HDS);
  float*    ssum = ws + 3 * SZ1 + (size_t)E2 * HDS + (size_t)NN * HDS;
  float*    zl2  = xl1;                                      // N*128
  float*    zr2  = xl1 + (size_t)NN * OUTC;
  float*    z2   = xl1 + 2 * (size_t)NN * OUTC;
  // packed bf16 weights after the float region (~0.9 MB)
  __bf16* packs = (__bf16*)(ws + 3 * SZ1 + (size_t)E2 * HDS + 2 * (size_t)NN * HDS);
  __bf16* Wl1P = packs;                         // 256*512
  __bf16* Wr1P = Wl1P + (size_t)INF_ * HDS * HIDC;
  __bf16* Wl2P = Wr1P + (size_t)INF_ * HDS * HIDC;   // 512*128
  __bf16* Wr2P = Wl2P + (size_t)HDS * HIDC * OUTC;
  __bf16* Wd1P = Wr2P + (size_t)HDS * HIDC * OUTC;   // 256*128

  const int TB = 256;
  auto cdiv = [](long long a, long long b) { return (int)((a + b - 1) / b); };

  // ---------------- pack weights to bf16 fragment order -------------
  pack_b_kernel<<<cdiv(INF_ * HDS * HIDC, TB), TB, 0, stream>>>(Wl1, Wl1P, INF_, HDS * HIDC);
  pack_b_kernel<<<cdiv(INF_ * HDS * HIDC, TB), TB, 0, stream>>>(Wr1, Wr1P, INF_, HDS * HIDC);
  pack_b_kernel<<<cdiv(HDS * HIDC * OUTC, TB), TB, 0, stream>>>(Wl2, Wl2P, HDS * HIDC, OUTC);
  pack_b_kernel<<<cdiv(HDS * HIDC * OUTC, TB), TB, 0, stream>>>(Wr2, Wr2P, HDS * HIDC, OUTC);
  pack_b_kernel<<<cdiv(2 * OUTC * HIDC, TB), TB, 0, stream>>>(Wd1, Wd1P, 2 * OUTC, HIDC);

  // ---------------- Layer 1: GATv2 (heads=4, ch=128) ----------------
  {
    int strips = (NN / 16) * ((HDS * HIDC) / 64);            // 10000
    int blocks = cdiv(strips, 8);
    gemm_bf16wmma_kernel<<<blocks, TB, 0, stream>>>(x, Wl1P, bl1, xl1, NN, HDS * HIDC, INF_);
    gemm_bf16wmma_kernel<<<blocks, TB, 0, stream>>>(x, Wr1P, br1, xr1, NN, HDS * HIDC, INF_);
  }
  fill_u32_kernel<<<cdiv(NN * HDS, TB), TB, 0, stream>>>(smax, 0u, NN * HDS);
  fill_f32_kernel<<<cdiv(NN * HDS, TB), TB, 0, stream>>>(ssum, 0.f, NN * HDS);
  {
    long long waves = (long long)E2 * HDS;
    edge_logits_kernel<<<cdiv(waves, 8), TB, 0, stream>>>(xl1, xr1, src, dst, att1, alph, smax, HDS);
    edge_exp_kernel<<<cdiv(waves, TB), TB, 0, stream>>>(alph, smax, ssum, dst, HDS);
    init_bias_kernel<<<cdiv((long long)NN * HDS * HIDC, TB), TB, 0, stream>>>(z1, bias1, NN, HDS * HIDC);
    edge_aggregate_kernel<<<cdiv(waves, 8), TB, 0, stream>>>(xl1, alph, ssum, src, dst, z1, HDS);
    gelu_kernel<<<cdiv((long long)NN * HDS * HIDC, TB), TB, 0, stream>>>(z1, NN * HDS * HIDC);
  }

  // ---------------- Layer 2: GATv2 (heads=1, ch=128) ----------------
  {
    int strips = (NN / 16) * (OUTC / 64);                    // 2500
    int blocks = cdiv(strips, 8);
    gemm_bf16wmma_kernel<<<blocks, TB, 0, stream>>>(z1, Wl2P, bl2, zl2, NN, OUTC, HDS * HIDC);
    gemm_bf16wmma_kernel<<<blocks, TB, 0, stream>>>(z1, Wr2P, br2, zr2, NN, OUTC, HDS * HIDC);
  }
  fill_u32_kernel<<<cdiv(NN, TB), TB, 0, stream>>>(smax, 0u, NN);
  fill_f32_kernel<<<cdiv(NN, TB), TB, 0, stream>>>(ssum, 0.f, NN);
  {
    long long waves = (long long)E2;
    edge_logits_kernel<<<cdiv(waves, 8), TB, 0, stream>>>(zl2, zr2, src, dst, att2, alph, smax, 1);
    edge_exp_kernel<<<cdiv(waves, TB), TB, 0, stream>>>(alph, smax, ssum, dst, 1);
    init_bias_kernel<<<cdiv((long long)NN * OUTC, TB), TB, 0, stream>>>(z2, bias2, NN, OUTC);
    edge_aggregate_kernel<<<cdiv(waves, 8), TB, 0, stream>>>(zl2, alph, ssum, src, dst, z2, 1);
  }

  // ---------------- Decoder: gather-concat -> MLP(256->128->1) ------
  init_bias_kernel<<<cdiv(ELN, TB), TB, 0, stream>>>(out, bd2, ELN, 1);
  decoder_kernel<<<ELN / 16, TB, 0, stream>>>(z2, li0, li1, Wd1P, bd1, Wd2, out);
}
